// MRTE_4088808866366
// MI455X (gfx1250) — matrix operand order repaired and checked by reading, not compiled, not run on hardware
//
#include <hip/hip_runtime.h>

// ---------------------------------------------------------------------------
// Problem constants (from the reference)
// ---------------------------------------------------------------------------
constexpr int Bv = 4, CT = 192, Cv = 768, Tv = 2048, Sv = 1024, Kv = 12;

typedef unsigned short u16;
typedef unsigned int   u32;
typedef __attribute__((ext_vector_type(16))) __bf16 v16bf;
typedef __attribute__((ext_vector_type(8)))  float  v8f;

union BF16x16 { v16bf v; u32 u[8]; };

__device__ __forceinline__ u16 f2bf(float x) {
  u32 u = __float_as_uint(x);
  u32 r = u + 0x7FFFu + ((u >> 16) & 1u);   // round-to-nearest-even
  return (u16)(r >> 16);
}
__device__ __forceinline__ u32 pack2bf(float a, float b) {
  return (u32)f2bf(a) | ((u32)f2bf(b) << 16);
}

// ---------------------------------------------------------------------------
// Prep kernels
// ---------------------------------------------------------------------------
__global__ __launch_bounds__(256) void k_prep_y(const float* __restrict__ y,
                                                u16* __restrict__ yT) {
  int i = blockIdx.x * 256 + threadIdx.x;           // over B*C*T
  int b = i / (Cv * Tv);
  int r = i - b * (Cv * Tv);
  int c = r / Tv;
  int t = r - c * Tv;
  yT[((size_t)b * Tv + t) * Cv + c] = f2bf(y[i]);
}

__global__ __launch_bounds__(256) void k_prep_text(const float* __restrict__ text,
                                                   u16* __restrict__ textb,
                                                   u16* __restrict__ textT) {
  int i = blockIdx.x * 256 + threadIdx.x;           // over B*CT*S
  int b = i / (CT * Sv);
  int r = i - b * (CT * Sv);
  int c = r / Sv;
  int s = r - c * Sv;
  u16 v = f2bf(text[i]);
  textb[i] = v;
  textT[((size_t)b * Sv + s) * CT + c] = v;
}

__global__ __launch_bounds__(256) void k_prep_w(const float* __restrict__ W,
                                                u16* __restrict__ wb) {
  int i = blockIdx.x * 256 + threadIdx.x;           // over K*C*CT
  wb[i] = f2bf(W[i]);
}

// ---------------------------------------------------------------------------
// tp[k,b,s,c] = text_mask[b,s] * scale[k] * (sum_j W[k,c,j]*text[b,j,s] + bias[k,c])
// ---------------------------------------------------------------------------
__global__ __launch_bounds__(32) void k_tp(const u16* __restrict__ textT,
                                           const u16* __restrict__ wb,
                                           const float* __restrict__ bias,
                                           const float* __restrict__ scale,
                                           const float* __restrict__ text_mask,
                                           u16* __restrict__ tp) {
  const int lane = threadIdx.x;
  const int hf = lane >> 4;
  const int lx = lane & 15;
  int bi = blockIdx.x;
  const int ct = bi % (Cv / 16); bi /= (Cv / 16);
  const int st = bi % (Sv / 16); bi /= (Sv / 16);
  const int bb = bi % Bv;        bi /= Bv;
  const int kh = bi;
  const int c0 = ct * 16, s0 = st * 16;

  const u16* tA = textT + ((size_t)bb * Sv + s0 + lx) * CT;      // A row: s
  const u16* tB = wb + ((size_t)kh * Cv + c0 + lx) * CT;         // B^T row: c

  v8f d = {};
#pragma unroll
  for (int ch = 0; ch < 6; ++ch) {
    const int j0 = ch * 32;
    BF16x16 a, b;
#pragma unroll
    for (int v = 0; v < 8; ++v) {
      const int ka = j0 + ((v & 3) * 2 + hf * 8 + (v >> 2) * 16);  // A K-pattern
      a.u[v] = *(const u32*)(tA + ka);
      const int kb = j0 + ((v & 3) * 2 + hf * 8 + (v >> 2) * 16);  // B K-pattern
      b.u[v] = *(const u32*)(tB + kb);
    }
    d = __builtin_amdgcn_wmma_f32_16x16x32_bf16(false, a.v, false, b.v,
                                                (short)0, d, false, false);
  }

  const float sc = scale[kh];
  const float bc = bias[kh * Cv + c0 + lx];
  u16* outp = tp + ((size_t)(kh * Bv + bb) * Sv) * Cv;
#pragma unroll
  for (int v = 0; v < 8; ++v) {
    const int srow = s0 + v + hf * 8;                // D: M = vgpr + 8*half
    float val = (d[v] + bc) * sc * text_mask[bb * Sv + srow];
    outp[(size_t)srow * Cv + c0 + lx] = f2bf(val);
  }
}

// ---------------------------------------------------------------------------
// Fused attention. Block = 4 waves (128 threads); each wave owns one 16-row
// t-tile, all waves share K-tiles staged in LDS by double-buffered
// global_load_async_to_lds_b128 (ASYNCcnt path).
// Dynamic LDS layout:
//   [0)        Q:   4 waves * 16x768 bf16   = 98304 B
//   [98304)    K:   2 bufs  * 32x768 bf16   = 98304 B
//   [196608)   P:   4 waves * 16x33 f32     =  8448 B
// ---------------------------------------------------------------------------
constexpr u32 QLDS_BYTES = 4 * 16 * Cv * 2;          // 98304
constexpr u32 KLDS_BYTES = 2 * 32 * Cv * 2;          // 98304
constexpr u32 PLDS_BYTES = 4 * 16 * 33 * 4;          //  8448
constexpr u32 LDS_TOTAL  = QLDS_BYTES + KLDS_BYTES + PLDS_BYTES;

__global__ __launch_bounds__(128) void k_attn(const u16* __restrict__ yT,
                                              const u16* __restrict__ tp,
                                              const u16* __restrict__ textb,
                                              const float* __restrict__ y_mask,
                                              const float* __restrict__ text_mask,
                                              const float* __restrict__ y,
                                              const float* __restrict__ ge,
                                              float* __restrict__ out) {
  extern __shared__ char smem[];
  u16*   qall = (u16*)smem;
  u16*   kst  = (u16*)(smem + QLDS_BYTES);
  float* pall = (float*)(smem + QLDS_BYTES + KLDS_BYTES);

  const int tid = threadIdx.x;
  const int w = tid >> 5;          // wave in block
  const int lane = tid & 31;
  const int hf = lane >> 4;
  const int lx = lane & 15;
  const int t0 = (blockIdx.x * 4 + w) * 16;
  const int bb = blockIdx.y;

  u16* qw = qall + (size_t)w * 16 * Cv;
  float (*plds)[33] = (float(*)[33])(pall + (size_t)w * 16 * 33);

  // Stage this wave's Q strip (contiguous 16 rows of yT)
  {
    const u32* qsrc = (const u32*)(yT + ((size_t)bb * Tv + t0) * Cv);
    u32* qdst = (u32*)qw;
    for (int i = lane; i < 16 * Cv / 2; i += 32) qdst[i] = qsrc[i];
  }

  float ym[8];
#pragma unroll
  for (int v = 0; v < 8; ++v) ym[v] = y_mask[bb * Tv + t0 + v + hf * 8];

  v8f acc[12];
#pragma unroll
  for (int c = 0; c < 12; ++c) acc[c] = (v8f){};

  const float rsq = 0.03608439182435161f;  // 1/sqrt(768)
  const u16* qrow = qw + (size_t)lx * Cv;

  // Async-stage 32 K-rows (48 KB contiguous) into LDS buffer `buf`;
  // 128 threads x 24 x 16B. Tracked by ASYNCcnt.
  auto issueK = [&](const u16* kbase, int s0, int buf) {
    const char* src = (const char*)(kbase + (size_t)s0 * Cv);
    const u32 ldsbase = (u32)(uintptr_t)(kst + (size_t)buf * 32 * Cv);
    for (int j = 0; j < 24; ++j) {
      const u32 off = (u32)((tid + j * 128) * 16);
      const u32 ldsa = ldsbase + off;
      asm volatile("global_load_async_to_lds_b128 %0, %1, %2"
                   :: "v"(ldsa), "v"(off), "s"(src) : "memory");
    }
  };
#define WAIT_ASYNC() asm volatile("s_wait_asynccnt 0" ::: "memory")

  // scores for the two 16-s tiles of LDS buffer `buf` (K-dim 768 = 24 WMMAs each)
  auto scores = [&](int buf, v8f& d0, v8f& d1) {
    const u16* kl = kst + (size_t)buf * 32 * Cv;
    const u16* krow0 = kl + (size_t)lx * Cv;          // local s rows 0..15
    const u16* krow1 = kl + (size_t)(16 + lx) * Cv;   // local s rows 16..31
#pragma unroll 4
    for (int ch = 0; ch < 24; ++ch) {
      const int cb = ch * 32;
      BF16x16 a, b0, b1;
#pragma unroll
      for (int v = 0; v < 8; ++v) {
        const int ka = cb + ((v & 3) * 2 + hf * 8 + (v >> 2) * 16);
        a.u[v] = *(const u32*)(qrow + ka);
        const int kb = cb + ((v & 3) * 2 + hf * 8 + (v >> 2) * 16);
        b0.u[v] = *(const u32*)(krow0 + kb);
        b1.u[v] = *(const u32*)(krow1 + kb);
      }
      d0 = __builtin_amdgcn_wmma_f32_16x16x32_bf16(false, a.v, false, b0.v,
                                                   (short)0, d0, false, false);
      d1 = __builtin_amdgcn_wmma_f32_16x16x32_bf16(false, a.v, false, b1.v,
                                                   (short)0, d1, false, false);
    }
  };

  for (int kh = 0; kh < Kv; ++kh) {
    const u16* kbase = tp + ((size_t)(kh * Bv + bb) * Sv) * Cv;

    // ---- pass 1: row max + denominator (online, exact) ----
    float m[8], l[8];
#pragma unroll
    for (int v = 0; v < 8; ++v) { m[v] = -1e30f; l[v] = 0.f; }

    __syncthreads();                 // buffers free (prev compute finished)
    issueK(kbase, 0, 0);
    for (int st = 0; st < 32; ++st) {
      WAIT_ASYNC();                  // my share of stage `st` arrived
      __syncthreads();               // everyone's share arrived
      if (st + 1 < 32) issueK(kbase, (st + 1) * 32, (st + 1) & 1);
      v8f d0 = {}, d1 = {};
      scores(st & 1, d0, d1);
      const int s0 = st * 32;
      const float tm0 = text_mask[bb * Sv + s0 + lx];
      const float tm1 = text_mask[bb * Sv + s0 + 16 + lx];
#pragma unroll
      for (int v = 0; v < 8; ++v) {
        const float a0 = d0[v] * ym[v] * tm0 * rsq;
        const float a1 = d1[v] * ym[v] * tm1 * rsq;
        float mx = fmaxf(a0, a1);
        mx = fmaxf(mx, __shfl_xor(mx, 1, 32));
        mx = fmaxf(mx, __shfl_xor(mx, 2, 32));
        mx = fmaxf(mx, __shfl_xor(mx, 4, 32));
        mx = fmaxf(mx, __shfl_xor(mx, 8, 32));
        const float m2 = fmaxf(m[v], mx);
        const float f = __expf(m[v] - m2);
        float rs = __expf(a0 - m2) + __expf(a1 - m2);
        rs += __shfl_xor(rs, 1, 32);
        rs += __shfl_xor(rs, 2, 32);
        rs += __shfl_xor(rs, 4, 32);
        rs += __shfl_xor(rs, 8, 32);
        l[v] = l[v] * f + rs;
        m[v] = m2;
      }
    }
    float invl[8];
#pragma unroll
    for (int v = 0; v < 8; ++v) invl[v] = 1.0f / l[v];

    // ---- pass 2: P = exp(s-m)/l, context += P @ V ----
    __syncthreads();
    issueK(kbase, 0, 0);
    for (int st = 0; st < 32; ++st) {
      WAIT_ASYNC();
      __syncthreads();
      if (st + 1 < 32) issueK(kbase, (st + 1) * 32, (st + 1) & 1);
      v8f d0 = {}, d1 = {};
      scores(st & 1, d0, d1);
      const int s0 = st * 32;
      const float tm0 = text_mask[bb * Sv + s0 + lx];
      const float tm1 = text_mask[bb * Sv + s0 + 16 + lx];
      __syncthreads();
#pragma unroll
      for (int v = 0; v < 8; ++v) {
        const float a0 = d0[v] * ym[v] * tm0 * rsq;
        const float a1 = d1[v] * ym[v] * tm1 * rsq;
        plds[v + hf * 8][lx]      = __expf(a0 - m[v]) * invl[v];
        plds[v + hf * 8][lx + 16] = __expf(a1 - m[v]) * invl[v];
      }
      __syncthreads();
      BF16x16 ap;   // D-layout -> A-layout transpose through LDS
#pragma unroll
      for (int v = 0; v < 8; ++v) {
        const int ka = (v & 3) * 2 + hf * 8 + (v >> 2) * 16;
        ap.u[v] = pack2bf(plds[lx][ka], plds[lx][ka + 1]);
      }
      const u16* vbase = textb + (size_t)bb * CT * Sv + s0;
#pragma unroll
      for (int c = 0; c < 12; ++c) {
        BF16x16 bv;
#pragma unroll
        for (int v = 0; v < 8; ++v)
          bv.u[v] = *(const u32*)(vbase + (size_t)(c * 16 + lx) * Sv + (v & 3) * 2 + hf * 8 + (v >> 2) * 16);
        acc[c] = __builtin_amdgcn_wmma_f32_16x16x32_bf16(false, ap.v, false, bv.v,
                                                         (short)0, acc[c], false, false);
      }
    }
  }

  // ---- epilogue: out[b,c,t] = y + ge + ctx[c % 192], c tiled 4x ----
  for (int rep = 0; rep < Cv / CT; ++rep) {
#pragma unroll
    for (int c = 0; c < 12; ++c) {
#pragma unroll
      for (int v = 0; v < 8; ++v) {
        const int cfull = rep * CT + c * 16 + lx;
        const int trow = t0 + v + hf * 8;
        const size_t idx = ((size_t)bb * Cv + cfull) * Tv + trow;
        out[idx] = y[idx] + ge[bb * Cv + cfull] + acc[c][v];
      }
    }
  }
}

// ---------------------------------------------------------------------------
extern "C" void kernel_launch(void* const* d_in, const int* in_sizes, int n_in,
                              void* d_out, int out_size, void* d_ws, size_t ws_size,
                              hipStream_t stream) {
  const float* y         = (const float*)d_in[0];
  const float* y_mask    = (const float*)d_in[1];
  const float* text      = (const float*)d_in[2];
  const float* text_mask = (const float*)d_in[3];
  const float* ge        = (const float*)d_in[4];
  const float* W         = (const float*)d_in[5];
  const float* bias      = (const float*)d_in[6];
  const float* scale     = (const float*)d_in[7];
  float* out = (float*)d_out;

  char* ws = (char*)d_ws;
  size_t off = 0;
  auto carve = [&](size_t bytes) { void* p = ws + off; off += (bytes + 255) & ~size_t(255); return p; };
  u16* yT    = (u16*)carve((size_t)Bv * Tv * Cv * 2);      // 12.6 MB
  u16* textb = (u16*)carve((size_t)Bv * CT * Sv * 2);      //  1.6 MB
  u16* textT = (u16*)carve((size_t)Bv * Sv * CT * 2);      //  1.6 MB
  u16* wb    = (u16*)carve((size_t)Kv * Cv * CT * 2);      //  3.5 MB
  u16* tpb   = (u16*)carve((size_t)Kv * Bv * Sv * Cv * 2); // 75.5 MB

  k_prep_y   <<<(Bv * Cv * Tv) / 256, 256, 0, stream>>>(y, yT);
  k_prep_text<<<(Bv * CT * Sv) / 256, 256, 0, stream>>>(text, textb, textT);
  k_prep_w   <<<(Kv * Cv * CT) / 256, 256, 0, stream>>>(W, wb);

  k_tp<<<Kv * Bv * (Sv / 16) * (Cv / 16), 32, 0, stream>>>(textT, wb, bias, scale,
                                                           text_mask, tpb);

  dim3 grid(Tv / 64, Bv);
  k_attn<<<grid, 128, LDS_TOTAL, stream>>>(yT, tpb, textb, y_mask, text_mask,
                                           y, ge, out);
}
